// RNN_6451040879094
// MI455X (gfx1250) — compile-verified, matrix operand order checked by reading, not run on hardware
//
#include <hip/hip_runtime.h>
#include <hip/hip_bf16.h>
#include <math.h>

// ---------------------------------------------------------------------------
// Types for gfx1250 WMMA (wave32): A/B = 16 x bf16 per lane, C/D = 8 x f32.
// ---------------------------------------------------------------------------
typedef __attribute__((ext_vector_type(16))) __bf16 v16bf;
typedef __attribute__((ext_vector_type(8)))  __bf16 bf16x8;
typedef __attribute__((ext_vector_type(4)))  __bf16 bf16x4;
typedef __attribute__((ext_vector_type(8)))  float  v8f;
typedef __attribute__((ext_vector_type(4)))  float  f32x4;

#define BATCH   4096
#define HSIZE   1024
#define OSIZE   4096
#define CATK    2048

// Block tile geometry: 256 threads = 8 waves, wave grid 2(M) x 4(N),
// wave tile 64x64 -> block tile 128(M) x 256(N), K-step 32.
#define BM      128
#define BN      256
#define BK      32

// ---------------------------------------------------------------------------
// CDNA5 async copy: global -> LDS, 16B per lane, tracked by ASYNCcnt.
// lds_off is the 32-bit LDS byte address (low 32 bits of a generic pointer
// to __shared__), gaddr is the 64-bit global address (VGPR pair).
// ---------------------------------------------------------------------------
__device__ __forceinline__ void async_cp16(const void* g, unsigned lds_off) {
  asm volatile("global_load_async_to_lds_b128 %0, %1, off"
               :: "v"(lds_off), "v"((unsigned long long)(uintptr_t)g)
               : "memory");
}

__device__ __forceinline__ void wait_async0() {
#if __has_builtin(__builtin_amdgcn_s_wait_asynccnt)
  __builtin_amdgcn_s_wait_asynccnt(0);
#else
  asm volatile("s_wait_asynccnt 0" ::: "memory");
#endif
}

// ---------------------------------------------------------------------------
// Elementwise f32 -> bf16 conversion, 4 elements / thread.
// n must be a multiple of 1024.
// ---------------------------------------------------------------------------
__global__ void __launch_bounds__(256)
cvt_bf16_kernel(const float* __restrict__ in, __bf16* __restrict__ out)
{
  const size_t i = (size_t)blockIdx.x * 256 + threadIdx.x;
  f32x4 f = reinterpret_cast<const f32x4*>(in)[i];
  bf16x4 b;
#pragma unroll
  for (int j = 0; j < 4; ++j) b[j] = (__bf16)f[j];
  reinterpret_cast<bf16x4*>(out)[i] = b;
}

// ---------------------------------------------------------------------------
// Load one 16x32 bf16 fragment for this lane from an LDS tile row.
// Layout per CDNA5 ISA 7.12.2 (16-bit A-matrix 16x32):
//   lanes 0-15 : K = {k0..k0+7} U {k0+16..k0+23}
//   lanes 16-31: K = {k0+8..k0+15} U {k0+24..k0+31}
// `rp` points at this lane's row (32 bf16); `k` = half-wave offset (0 or 8).
// Two ds_load_b128 per fragment.
// ---------------------------------------------------------------------------
__device__ __forceinline__ v16bf frag_ld_lds(const __bf16* rp, int k) {
  bf16x8 lo = *reinterpret_cast<const bf16x8*>(rp + k);
  bf16x8 hi = *reinterpret_cast<const bf16x8*>(rp + k + 16);
  return __builtin_shufflevector(lo, hi, 0, 1, 2, 3, 4, 5, 6, 7,
                                 8, 9, 10, 11, 12, 13, 14, 15);
}

// ---------------------------------------------------------------------------
// Issue async loads for one KxROWS bf16 tile (ROWS x 32 elems, 16B chunks).
// ROWS*4 chunks striped over 256 threads (uniform trip count, EXEC full).
// ---------------------------------------------------------------------------
template <int ROWS>
__device__ __forceinline__ void issue_tile(const __bf16* __restrict__ G, int ld,
                                           int rowbase, int k,
                                           __bf16* lds, int tid)
{
#pragma unroll
  for (int c = 0; c < ROWS * 4 / 256; ++c) {
    const int chunk = tid + c * 256;
    const int r   = chunk >> 2;
    const int off = (chunk & 3) * 8;   // bf16 elems (8 = 16B)
    const __bf16* g = G + (size_t)(rowbase + r) * ld + k + off;
    async_cp16(g, (unsigned)(uintptr_t)(lds + r * BK + off));
  }
}

// ---------------------------------------------------------------------------
// One K-sweep of acc += A * W^T for the whole block, LDS-staged and
// double-buffered with async global->LDS copies.
// A: [M x K] bf16 row-major, W: [N x K] bf16 row-major.
// ---------------------------------------------------------------------------
__device__ __forceinline__ void gemm_part_staged(
    const __bf16* __restrict__ A, int lda,
    const __bf16* __restrict__ W, int ldw, int K,
    int mblk, int nblk, int tid, int wave, int lr, int lh,
    __bf16* lA, __bf16* lB, v8f acc[4][4])
{
  const int nsteps = K / BK;
  const int arow = (wave & 1) * 64 + lr;   // this lane's A rows base-in-tile
  const int brow = (wave >> 1) * 64 + lr;  // this lane's B rows base-in-tile

  // Prologue: fill buffer 0.
  issue_tile<BM>(A, lda, mblk, 0, lA, tid);
  issue_tile<BN>(W, ldw, nblk, 0, lB, tid);
  wait_async0();
  __syncthreads();

  for (int s = 0; s < nsteps; ++s) {
    const int buf = s & 1;
    const int nxt = buf ^ 1;
    if (s + 1 < nsteps) {
      issue_tile<BM>(A, lda, mblk, (s + 1) * BK, lA + nxt * (BM * BK), tid);
      issue_tile<BN>(W, ldw, nblk, (s + 1) * BK, lB + nxt * (BN * BK), tid);
    }

    const __bf16* tA = lA + buf * (BM * BK);
    const __bf16* tB = lB + buf * (BN * BK);
    v16bf fa[4], fb[4];
#pragma unroll
    for (int i = 0; i < 4; ++i) {
      fa[i] = frag_ld_lds(tA + (arow + i * 16) * BK, lh);
      fb[i] = frag_ld_lds(tB + (brow + i * 16) * BK, lh);
    }
#pragma unroll
    for (int mi = 0; mi < 4; ++mi)
#pragma unroll
      for (int t = 0; t < 4; ++t)
        acc[mi][t] = __builtin_amdgcn_wmma_f32_16x16x32_bf16(
            false, fa[mi], false, fb[t], (short)0, acc[mi][t], false, false);

    if (s + 1 < nsteps) wait_async0();
    __syncthreads();
  }
}

// ---------------------------------------------------------------------------
// C = act( A1*W1^T + A2*W2^T + b1 (+ b2) ), act = tanh or identity.
// grid.x = M/128, grid.y = N/256. Optionally emits a bf16 copy of C.
// ---------------------------------------------------------------------------
template <bool TANH>
__global__ void __launch_bounds__(256)
dual_gemm_bf16(const __bf16* __restrict__ A1, int lda1,
               const __bf16* __restrict__ W1, int ldw1,
               const __bf16* __restrict__ A2, int lda2,
               const __bf16* __restrict__ W2, int ldw2,
               const float* __restrict__ b1,
               const float* __restrict__ b2,
               float* __restrict__ C,
               __bf16* __restrict__ Cbf,
               int ldc, int K1, int K2)
{
  __shared__ __bf16 lA[2 * BM * BK];   //  16 KB
  __shared__ __bf16 lB[2 * BN * BK];   //  32 KB

  const int tid  = threadIdx.x;
  const int lane = tid & 31;
  const int wave = tid >> 5;
  const int lr   = lane & 15;          // row/col within a 16-tile
  const int lh   = (lane >> 4) << 3;   // half-wave K offset (0 or 8)

  const int mblk = blockIdx.x * BM;
  const int nblk = blockIdx.y * BN;
  const int row0 = mblk + (wave & 1) * 64;
  const int col0 = nblk + (wave >> 1) * 64;

  v8f acc[4][4] = {};

  gemm_part_staged(A1, lda1, W1, ldw1, K1, mblk, nblk, tid, wave, lr, lh, lA, lB, acc);
  gemm_part_staged(A2, lda2, W2, ldw2, K2, mblk, nblk, tid, wave, lr, lh, lA, lB, acc);

  // Epilogue per C/D layout: acc[mi][t][r] -> C[row0+mi*16+lh+r][col0+t*16+lr]
#pragma unroll
  for (int t = 0; t < 4; ++t) {
    const int col = col0 + t * 16 + lr;
    float bias = b1[col];
    if (b2) bias += b2[col];
#pragma unroll
    for (int mi = 0; mi < 4; ++mi) {
      const size_t base = (size_t)(row0 + mi * 16 + lh) * ldc + col;
#pragma unroll
      for (int r = 0; r < 8; ++r) {
        float v = acc[mi][t][r] + bias;
        if (TANH) v = tanhf(v);
        C[base + (size_t)r * ldc] = v;
        if (Cbf) Cbf[base + (size_t)r * ldc] = (__bf16)v;
      }
    }
  }
}

// ---------------------------------------------------------------------------
// In-place row-wise log_softmax. One block per row, N = 4096, 256 threads,
// 16 elements per thread in registers; two LDS tree reductions.
// ---------------------------------------------------------------------------
__global__ void __launch_bounds__(256)
logsoftmax_kernel(float* __restrict__ data, int N)
{
  float* row = data + (size_t)blockIdx.x * N;
  const int tid = threadIdx.x;

  float loc[16];
  float mx = -INFINITY;
#pragma unroll
  for (int i = 0; i < 16; ++i) {
    loc[i] = row[tid + i * 256];
    mx = fmaxf(mx, loc[i]);
  }

  __shared__ float red[256];
  red[tid] = mx;
  __syncthreads();
#pragma unroll
  for (int s = 128; s > 0; s >>= 1) {
    if (tid < s) red[tid] = fmaxf(red[tid], red[tid + s]);
    __syncthreads();
  }
  mx = red[0];
  __syncthreads();

  float sum = 0.0f;
#pragma unroll
  for (int i = 0; i < 16; ++i) sum += expf(loc[i] - mx);
  red[tid] = sum;
  __syncthreads();
#pragma unroll
  for (int s = 128; s > 0; s >>= 1) {
    if (tid < s) red[tid] += red[tid + s];
    __syncthreads();
  }
  const float lse = mx + logf(red[0]);

#pragma unroll
  for (int i = 0; i < 16; ++i) row[tid + i * 256] = loc[i] - lse;
}

// ---------------------------------------------------------------------------
// d_in order: input, hidden, W_ih0, W_hh0, b_ih0, b_hh0,
//             W_ih1, W_hh1, b_ih1, b_hh1, W_out, b_out
// d_out: [ logsoftmax 4096x4096 | h0 4096x1024 | h1 4096x1024 ]
// d_ws : bf16 copies (x, h0, hprev0, hprev1, 4 layer weights, W_out) ~56 MB
// ---------------------------------------------------------------------------
extern "C" void kernel_launch(void* const* d_in, const int* in_sizes, int n_in,
                              void* d_out, int out_size, void* d_ws, size_t ws_size,
                              hipStream_t stream) {
  (void)in_sizes; (void)n_in; (void)out_size; (void)ws_size;

  const float* input  = (const float*)d_in[0];   // [1, B, 1024]
  const float* hidden = (const float*)d_in[1];   // [2, B, 1024]
  const float* W_ih0  = (const float*)d_in[2];
  const float* W_hh0  = (const float*)d_in[3];
  const float* b_ih0  = (const float*)d_in[4];
  const float* b_hh0  = (const float*)d_in[5];
  const float* W_ih1  = (const float*)d_in[6];
  const float* W_hh1  = (const float*)d_in[7];
  const float* b_ih1  = (const float*)d_in[8];
  const float* b_hh1  = (const float*)d_in[9];
  const float* W_out  = (const float*)d_in[10];  // [4096, 2048]
  const float* b_out  = (const float*)d_in[11];

  float* out = (float*)d_out;                    // logits -> log_softmax
  float* h0  = out + (size_t)BATCH * OSIZE;      // new_hidden[0] (f32)
  float* h1  = h0 + (size_t)BATCH * HSIZE;       // new_hidden[1] (f32)
  const float* hprev0 = hidden;
  const float* hprev1 = hidden + (size_t)BATCH * HSIZE;

  // bf16 workspace layout
  __bf16* ws    = (__bf16*)d_ws;
  __bf16* xbf   = ws;                                  // 4096*1024
  __bf16* h0bf  = xbf   + (size_t)BATCH * HSIZE;       // 4096*1024
  __bf16* hp0bf = h0bf  + (size_t)BATCH * HSIZE;       // 4096*1024
  __bf16* hp1bf = hp0bf + (size_t)BATCH * HSIZE;       // 4096*1024
  __bf16* Wih0b = hp1bf + (size_t)BATCH * HSIZE;       // 1024*1024
  __bf16* Whh0b = Wih0b + (size_t)HSIZE * HSIZE;
  __bf16* Wih1b = Whh0b + (size_t)HSIZE * HSIZE;
  __bf16* Whh1b = Wih1b + (size_t)HSIZE * HSIZE;
  __bf16* Woutb = Whh1b + (size_t)HSIZE * HSIZE;       // 4096*2048

  dim3 blk(256);
  auto cvt = [&](const float* src, __bf16* dst, size_t n) {
    cvt_bf16_kernel<<<dim3((unsigned)(n / 1024)), blk, 0, stream>>>(src, dst);
  };

  // One-time f32 -> bf16 conversions (weights + activations).
  cvt(input,  xbf,   (size_t)BATCH * HSIZE);
  cvt(hprev0, hp0bf, (size_t)BATCH * HSIZE);
  cvt(hprev1, hp1bf, (size_t)BATCH * HSIZE);
  cvt(W_ih0,  Wih0b, (size_t)HSIZE * HSIZE);
  cvt(W_hh0,  Whh0b, (size_t)HSIZE * HSIZE);
  cvt(W_ih1,  Wih1b, (size_t)HSIZE * HSIZE);
  cvt(W_hh1,  Whh1b, (size_t)HSIZE * HSIZE);
  cvt(W_out,  Woutb, (size_t)OSIZE * CATK);

  // Layer 0: h0 = tanh(x W_ih0^T + hprev0 W_hh0^T + b_ih0 + b_hh0)
  dual_gemm_bf16<true><<<dim3(BATCH / BM, HSIZE / BN), blk, 0, stream>>>(
      xbf, HSIZE, Wih0b, HSIZE, hp0bf, HSIZE, Whh0b, HSIZE,
      b_ih0, b_hh0, h0, h0bf, HSIZE, HSIZE, HSIZE);

  // Layer 1: h1 = tanh(h0 W_ih1^T + hprev1 W_hh1^T + b_ih1 + b_hh1)
  dual_gemm_bf16<true><<<dim3(BATCH / BM, HSIZE / BN), blk, 0, stream>>>(
      h0bf, HSIZE, Wih1b, HSIZE, hp1bf, HSIZE, Whh1b, HSIZE,
      b_ih1, b_hh1, h1, (__bf16*)nullptr, HSIZE, HSIZE, HSIZE);

  // Output: logits = [x | h0] W_out^T + b_out (split K: cols 0..1023 / 1024..2047)
  dual_gemm_bf16<false><<<dim3(BATCH / BM, OSIZE / BN), blk, 0, stream>>>(
      xbf, HSIZE, Woutb, CATK, h0bf, HSIZE, Woutb + HSIZE, CATK,
      b_out, nullptr, out, (__bf16*)nullptr, OSIZE, HSIZE, HSIZE);

  // In-place row-wise log_softmax over 4096 columns.
  logsoftmax_kernel<<<dim3(BATCH), blk, 0, stream>>>(out, OSIZE);
}
